// PixelCrossAttentionRefiner_19782619365519
// MI455X (gfx1250) — matrix operand order repaired and checked by reading, not compiled, not run on hardware
//
#include <hip/hip_runtime.h>

// Problem constants (from the reference): B=2, K=8, C=E=128, H=W=128
#define HW 16384

typedef __attribute__((ext_vector_type(16))) _Float16 v16h;
typedef __attribute__((ext_vector_type(8)))  _Float16 v8h;
typedef __attribute__((ext_vector_type(8)))  float    v8f;
typedef __attribute__((ext_vector_type(4)))  float    v4f;

// ---------------------------------------------------------------------------
// Y[mapY][e][n] (+)= W[e, c0 : c0+128] @ X[mapY>>xshift][c][n]
// X, Y are channel-major [nMaps][128][HW] f32 feature maps.
// One workgroup = 8 wave32s; each wave owns 16 pixel columns, full M=128.
// Weight block staged to LDS as f16; math via v_wmma_f32_16x16x32_f16.
// In-place (Y==X) is safe: each wave reads only the columns it later writes.
// ---------------------------------------------------------------------------
__global__ __launch_bounds__(256) void gemm128_wmma(
    const float* __restrict__ X, const float* __restrict__ W,
    float* __restrict__ Y, int ldw, int c0, int xshift, int accFlag)
{
    __shared__ __attribute__((aligned(16))) _Float16 Wl[128 * 128];

    const int tid  = threadIdx.x;
    const int mapY = blockIdx.y;
    const int mapX = mapY >> xshift;
    const float* Xp = X + (size_t)mapX * 128 * HW;
    float*       Yp = Y + (size_t)mapY * 128 * HW;

    // Cooperative W -> LDS (f32 -> f16), 128x128 block at column offset c0.
    for (int p = tid; p < 128 * 64; p += 256) {
        int e = p >> 6;
        int c = (p & 63) << 1;
        Wl[e * 128 + c    ] = (_Float16)W[e * ldw + c0 + c    ];
        Wl[e * 128 + c + 1] = (_Float16)W[e * ldw + c0 + c + 1];
    }
    __syncthreads();

    const int wave   = tid >> 5;
    const int lane   = tid & 31;
    const int laneM  = lane & 15;   // A row within 16 / D column within 16
    const int laneHi = lane >> 4;   // selects K-half (A) / M+8 (D)
    const int ncol0  = blockIdx.x * 128 + wave * 16;

    v8f acc[8];
    if (accFlag) {
#pragma unroll
        for (int m = 0; m < 8; ++m)
#pragma unroll
            for (int r = 0; r < 8; ++r)
                acc[m][r] = Yp[(size_t)(16 * m + r + 8 * laneHi) * HW + ncol0 + laneM];
    } else {
#pragma unroll
        for (int m = 0; m < 8; ++m)
#pragma unroll
            for (int r = 0; r < 8; ++r)
                acc[m][r] = 0.0f;
    }

    const int khalf = laneHi * 8;   // A fragment: lanes 0-15 -> K {0..7,16..23}, lanes 16-31 -> +8
#pragma unroll
    for (int ks = 0; ks < 4; ++ks) {
        // B fragment: column = ncol0 + laneM; elems 0..7 -> k = khalf+0..7, 8..15 -> k = 16+khalf+0..7.
        const float* xc = Xp + (size_t)(ks * 32 + khalf) * HW + ncol0 + laneM;
        v16h bf;
#pragma unroll
        for (int i = 0; i < 8; ++i) {
            bf[i]     = (_Float16)xc[(size_t)i * HW];
            bf[8 + i] = (_Float16)xc[(size_t)(16 + i) * HW];
        }
#pragma unroll
        for (int m = 0; m < 8; ++m) {
            // A fragment: row = 16m + laneM; halves 0..7 -> k = khalf+0..7,
            // halves 8..15 -> k = 16+khalf+0..7  (ISA 16-bit A 16x32 layout).
            const _Float16* wrow = &Wl[(16 * m + laneM) * 128 + ks * 32 + khalf];
            v8h lo = *(const v8h*)(wrow);
            v8h hi = *(const v8h*)(wrow + 16);
            v16h af = __builtin_shufflevector(lo, hi,
                0, 1, 2, 3, 4, 5, 6, 7, 8, 9, 10, 11, 12, 13, 14, 15);
            acc[m] = __builtin_amdgcn_wmma_f32_16x16x32_f16(
                false, af, false, bf, (short)0, acc[m], false, false);
        }
    }

#pragma unroll
    for (int m = 0; m < 8; ++m)
#pragma unroll
        for (int r = 0; r < 8; ++r)
            Yp[(size_t)(16 * m + r + 8 * laneHi) * HW + ncol0 + laneM] = acc[m][r];
}

// ---------------------------------------------------------------------------
// Y[map][e][n] += bias[e] (+ W[e,ci]*i(n) + W[e,cj]*j(n))
// i(n) = (n>>7)/127, j(n) = (n&127)/127  (linspace(0,1,128))
// ---------------------------------------------------------------------------
__global__ __launch_bounds__(256) void bias_coord_epilogue(
    float* __restrict__ Y, const float* __restrict__ bias,
    const float* __restrict__ W, int ldw, int ci, int cj, int useCoords)
{
    const int idx = blockIdx.x * 256 + threadIdx.x;       // over 128*HW
    float* Yp = Y + (size_t)blockIdx.y * 128 * HW;
    const int e = idx >> 14;
    const int n = idx & (HW - 1);
    float v = Yp[idx] + bias[e];
    if (useCoords) {
        float fi = (float)(n >> 7)  * (1.0f / 127.0f);
        float fj = (float)(n & 127) * (1.0f / 127.0f);
        v += W[e * ldw + ci] * fi + W[e * ldw + cj] * fj;
    }
    Yp[idx] = v;
}

// ---------------------------------------------------------------------------
// Per query pixel p: candidate slot k' reads source map k = p>>11 and source
// pixel (p&2047)*8 + k'  (the torch permute/reshape reinterpretation).
// scores -> per-head softmax -> mean over heads -> L2 renormalize.
// AW layout: [p][k'] (8 contiguous weights per pixel).
// ---------------------------------------------------------------------------
__global__ __launch_bounds__(256) void attn_kernel(
    const float* __restrict__ QA,   // [128][HW]
    const float* __restrict__ KA,   // [8][128][HW]
    float* __restrict__ AW)         // [HW*8]
{
    const int p     = blockIdx.x * 256 + threadIdx.x;
    const int kmap  = p >> 11;
    const int qbase = (p & 2047) * 8;
    const float* kb = KA + (size_t)kmap * 128 * HW + qbase;

    float s[4][8];
#pragma unroll
    for (int h = 0; h < 4; ++h) {
#pragma unroll
        for (int k = 0; k < 8; ++k) s[h][k] = 0.0f;
#pragma unroll
        for (int d = 0; d < 32; ++d) {
            float qv = QA[(size_t)(h * 32 + d) * HW + p];
            const v4f* kr = (const v4f*)(kb + (size_t)(h * 32 + d) * HW);
            v4f k0 = kr[0], k1 = kr[1];
#pragma unroll
            for (int k = 0; k < 4; ++k) {
                s[h][k]     += qv * k0[k];
                s[h][k + 4] += qv * k1[k];
            }
        }
    }

    const float scale = 0.17677669529663687f;   // 1/sqrt(32)
    float aw[8];
#pragma unroll
    for (int k = 0; k < 8; ++k) aw[k] = 0.0f;
#pragma unroll
    for (int h = 0; h < 4; ++h) {
        float mx = -1e30f;
#pragma unroll
        for (int k = 0; k < 8; ++k) { s[h][k] *= scale; mx = fmaxf(mx, s[h][k]); }
        float sum = 0.0f;
#pragma unroll
        for (int k = 0; k < 8; ++k) { s[h][k] = __expf(s[h][k] - mx); sum += s[h][k]; }
        float inv = 1.0f / sum;
#pragma unroll
        for (int k = 0; k < 8; ++k) aw[k] += s[h][k] * inv;
    }
    float nrm2 = 0.0f;
#pragma unroll
    for (int k = 0; k < 8; ++k) { aw[k] *= 0.25f; nrm2 += aw[k] * aw[k]; }
    float inv = 1.0f / fmaxf(sqrtf(nrm2), 1e-12f);
#pragma unroll
    for (int k = 0; k < 8; ++k) AW[p * 8 + k] = aw[k] * inv;
}

// ---------------------------------------------------------------------------
// Out[c][p] = sum_k' AW[p][k'] * Z[p>>11][c][(p&2047)*8 + k']   (raw values)
// ---------------------------------------------------------------------------
__global__ __launch_bounds__(256) void wsum_kernel(
    const float* __restrict__ Z,    // [8][128][HW]
    const float* __restrict__ AW,   // [HW*8]
    float* __restrict__ Out)        // [128][HW]
{
    const int idx   = blockIdx.x * 256 + threadIdx.x;     // over 128*HW
    const int c     = idx >> 14;
    const int p     = idx & (HW - 1);
    const int kmap  = p >> 11;
    const int qbase = (p & 2047) * 8;
    const v4f* zr = (const v4f*)(Z + ((size_t)kmap * 128 + c) * HW + qbase);
    const v4f* ar = (const v4f*)(AW + p * 8);
    v4f z0 = zr[0], z1 = zr[1], a0 = ar[0], a1 = ar[1];
    float acc = 0.0f;
#pragma unroll
    for (int k = 0; k < 4; ++k) acc += a0[k] * z0[k] + a1[k] * z1[k];
    Out[idx] = acc;
}

// ---------------------------------------------------------------------------
extern "C" void kernel_launch(void* const* d_in, const int* in_sizes, int n_in,
                              void* d_out, int out_size, void* d_ws, size_t ws_size,
                              hipStream_t stream)
{
    (void)in_sizes; (void)n_in; (void)out_size; (void)ws_size;
    const float* HF_star  = (const float*)d_in[0];
    const float* HF_cands = (const float*)d_in[1];
    const float* Z_cands  = (const float*)d_in[2];
    const float* q1w  = (const float*)d_in[3];  const float* q1b  = (const float*)d_in[4];
    const float* k1w  = (const float*)d_in[5];  const float* k1b  = (const float*)d_in[6];
    const float* a1qw = (const float*)d_in[7];  const float* a1qb = (const float*)d_in[8];
    const float* a1kw = (const float*)d_in[9];  const float* a1kb = (const float*)d_in[10];
    const float* q2w  = (const float*)d_in[11]; const float* q2b  = (const float*)d_in[12];
    const float* k2w  = (const float*)d_in[13]; const float* k2b  = (const float*)d_in[14];
    const float* a2qw = (const float*)d_in[15]; const float* a2qb = (const float*)d_in[16];
    const float* a2kw = (const float*)d_in[17]; const float* a2kb = (const float*)d_in[18];

    float* out = (float*)d_out;                 // [z_star | z_hat], each [2][128][HW]
    const size_t MAP = (size_t)128 * HW;        // one feature map (f32 elems)

    // Workspace (~76 MB): bufQ [1 map] | bufK [8 maps] | bufW [HW*8]
    float* bufQ = (float*)d_ws;
    float* bufK = bufQ + MAP;
    float* bufW = bufK + 8 * MAP;

    const dim3 blk(256);
    for (int b = 0; b < 2; ++b) {
        const float* Xhf = HF_star  + (size_t)b * MAP;
        const float* Xcd = HF_cands + (size_t)b * 8 * MAP;
        const float* Xz  = Z_cands  + (size_t)b * 8 * MAP;
        float* zs = out + (size_t)b * MAP;              // z_star[b]
        float* zh = out + 2 * MAP + (size_t)b * MAP;    // z_hat[b]

        // ---------------- stage 1 ----------------
        gemm128_wmma<<<dim3(128, 1), blk, 0, stream>>>(Xhf, q1w, bufQ, 130, 0, 0, 0);
        bias_coord_epilogue<<<dim3(8192, 1), blk, 0, stream>>>(bufQ, q1b, q1w, 130, 128, 129, 1);
        gemm128_wmma<<<dim3(128, 8), blk, 0, stream>>>(Xcd, k1w, bufK, 130, 0, 0, 0);
        bias_coord_epilogue<<<dim3(8192, 8), blk, 0, stream>>>(bufK, k1b, k1w, 130, 128, 129, 1);
        gemm128_wmma<<<dim3(128, 1), blk, 0, stream>>>(bufQ, a1qw, bufQ, 128, 0, 0, 0);   // qa in-place
        bias_coord_epilogue<<<dim3(8192, 1), blk, 0, stream>>>(bufQ, a1qb, nullptr, 0, 0, 0, 0);
        gemm128_wmma<<<dim3(128, 8), blk, 0, stream>>>(bufK, a1kw, bufK, 128, 0, 0, 0);   // ka in-place
        bias_coord_epilogue<<<dim3(8192, 8), blk, 0, stream>>>(bufK, a1kb, nullptr, 0, 0, 0, 0);
        attn_kernel<<<dim3(64), blk, 0, stream>>>(bufQ, bufK, bufW);
        wsum_kernel<<<dim3(8192), blk, 0, stream>>>(Xz, bufW, zs);

        // ---------------- stage 2 ----------------
        gemm128_wmma<<<dim3(128, 1), blk, 0, stream>>>(Xhf, q2w, bufQ, 258, 0, 0, 0);
        gemm128_wmma<<<dim3(128, 1), blk, 0, stream>>>(zs,  q2w, bufQ, 258, 128, 0, 1);   // += z part
        bias_coord_epilogue<<<dim3(8192, 1), blk, 0, stream>>>(bufQ, q2b, q2w, 258, 256, 257, 1);
        gemm128_wmma<<<dim3(128, 8), blk, 0, stream>>>(Xcd, k2w, bufK, 258, 0, 0, 0);
        gemm128_wmma<<<dim3(128, 8), blk, 0, stream>>>(zs,  k2w, bufK, 258, 128, 3, 1);   // shared z part
        bias_coord_epilogue<<<dim3(8192, 8), blk, 0, stream>>>(bufK, k2b, k2w, 258, 256, 257, 1);
        gemm128_wmma<<<dim3(128, 1), blk, 0, stream>>>(bufQ, a2qw, bufQ, 128, 0, 0, 0);
        bias_coord_epilogue<<<dim3(8192, 1), blk, 0, stream>>>(bufQ, a2qb, nullptr, 0, 0, 0, 0);
        gemm128_wmma<<<dim3(128, 8), blk, 0, stream>>>(bufK, a2kw, bufK, 128, 0, 0, 0);
        bias_coord_epilogue<<<dim3(8192, 8), blk, 0, stream>>>(bufK, a2kb, nullptr, 0, 0, 0, 0);
        attn_kernel<<<dim3(64), blk, 0, stream>>>(bufQ, bufK, bufW);
        wsum_kernel<<<dim3(8192), blk, 0, stream>>>(Xz, bufW, zh);
    }
}
